// DCGRUCell_18030272708970
// MI455X (gfx1250) — matrix operand order repaired and checked by reading, not compiled, not run on hardware
//
#include <hip/hip_runtime.h>
#include <hip/hip_bf16.h>
#include <math.h>

// ---------------- Problem constants ----------------
#define NN_   4096            // N (nodes)
#define BB_   64              // B (batch)
#define UU_   64              // U (hidden)
#define FF_   66              // F = IN_DIM + U
#define FBL_  4224            // F * B   (columns of X)
#define KKR_  330             // F * NUM_MAT (real feature K)
#define KKP_  352             // padded to multiple of 32
#define ROWS_ 262144          // B * N   (rows of feature GEMM)

typedef _Float16 h_t;
typedef __attribute__((ext_vector_type(16))) _Float16 v16h;
typedef __attribute__((ext_vector_type(8)))  _Float16 v8h;
typedef __attribute__((ext_vector_type(4)))  _Float16 v4h;
typedef __attribute__((ext_vector_type(8)))  float    v8f;

static __device__ __forceinline__ v16h join8(v8h lo, v8h hi) {
    return __builtin_shufflevector(lo, hi, 0,1,2,3,4,5,6,7,8,9,10,11,12,13,14,15);
}

// ---------------- Pack kernels ----------------

// f32 -> f16 supports (2 x 4096 x 4096), 4 elements/thread
__global__ __launch_bounds__(256) void pack_supports(const float* __restrict__ s,
                                                     h_t* __restrict__ sh, int n4) {
    int i = blockIdx.x * 256 + threadIdx.x;
    if (i >= n4) return;
    float4 v = ((const float4*)s)[i];
    v4h o; o[0] = (h_t)v.x; o[1] = (h_t)v.y; o[2] = (h_t)v.z; o[3] = (h_t)v.w;
    *(v4h*)(sh + (size_t)i * 4) = o;
}

// W (KKR_ x O) f32 -> Wt (O x KKP_) f16, zero-padded K
__global__ __launch_bounds__(256) void pack_wt(const float* __restrict__ W,
                                               h_t* __restrict__ Wt, int O) {
    int i = blockIdx.x * 256 + threadIdx.x;
    if (i >= O * KKP_) return;
    int kk = i % KKP_, o = i / KKP_;
    Wt[i] = (kk < KKR_) ? (h_t)W[kk * O + o] : (h_t)0.0f;
}

// Build x0^T (FBL_ rows x NN_ cols) f16.  stage==1 uses state = r*hx (r = sigmoid gate in `value`).
__global__ __launch_bounds__(256) void pack_x0(const float* __restrict__ inputs,
                                               const float* __restrict__ hx,
                                               const float* __restrict__ value,
                                               int stage, h_t* __restrict__ x0t) {
    unsigned id = blockIdx.x * 256 + threadIdx.x;   // FBL_*NN_ = 17,301,504 threads
    int n = id & (NN_ - 1);
    int c = id >> 12;                 // 0..4223 ; c = f*64 + b
    int f = c >> 6, b = c & 63;
    float v;
    if (f < 2) {
        v = inputs[(size_t)b * (NN_ * 2) + n * 2 + f];
    } else {
        int o = f - 2;
        size_t r = (size_t)b * NN_ + n;
        float h = hx[r * UU_ + o];
        if (stage) h *= value[r * 128 + o];
        v = h;
    }
    x0t[(size_t)id] = (h_t)v;
}

// Gather Afeat (ROWS_ x KKP_) f16 from the 5 transposed Chebyshev buffers xsT[m] (FBL_ x NN_ each).
__global__ __launch_bounds__(256) void pack_afeat(const h_t* __restrict__ xsT,
                                                  h_t* __restrict__ Af) {
    unsigned id = blockIdx.x * 256 + threadIdx.x;   // 64*352*4096 = 92,274,688 threads
    int n    = id & (NN_ - 1);
    int rest = id >> 12;              // b*KKP_ + kk
    int kk   = rest % KKP_;
    int b    = rest / KKP_;
    h_t v = (h_t)0.0f;
    if (kk < KKR_) {
        int m = kk % 5;               // stack order: x0, s0x1, s0x2, s1x1, s1x2
        int f = kk / 5;
        v = xsT[((size_t)m * FBL_ + f * 64 + b) * NN_ + n];
    }
    Af[((size_t)b * NN_ + n) * KKP_ + kk] = v;
}

// ---------------- Big diffusion GEMM:  Ct = (alpha * S@X + beta * E)^T ----------------
// A  : S   (4096 x 4096) row-major f16
// Bt : X^T (4224 x 4096) f16   (Bt[c][k] = X[k][c])
// Et : optional X0^T for Chebyshev epilogue
// Ct : out^T (4224 x 4096) f16
// Block = 8 waves = 4(M) x 2(C); wave tile 64(M) x 64(C); block tile 256 x 128.
__global__ __launch_bounds__(256) void wmma_sx_gemm(const h_t* __restrict__ A,
                                                    const h_t* __restrict__ Bt,
                                                    const h_t* __restrict__ Et,
                                                    h_t* __restrict__ Ct,
                                                    float alpha, float beta) {
    const int lane = threadIdx.x & 31;
    const int lh   = lane & 15;
    const int hp   = lane >> 4;
    const int wid  = threadIdx.x >> 5;
    const int m0   = blockIdx.x * 256 + (wid & 3) * 64;
    const int c0   = blockIdx.y * 128 + (wid >> 2) * 64;

    v8f acc[4][4] = {};
    for (int k0 = 0; k0 < NN_; k0 += 32) {
        v16h af[4];
#pragma unroll
        for (int t = 0; t < 4; ++t) {
            const h_t* p = A + (size_t)(m0 + t * 16 + lh) * NN_ + k0 + hp * 8;
            af[t] = join8(*(const v8h*)p, *(const v8h*)(p + 16));
        }
        v16h bf[4];
#pragma unroll
        for (int u = 0; u < 4; ++u) {
            const h_t* p = Bt + (size_t)(c0 + u * 16 + lh) * NN_ + k0 + hp * 8;
            bf[u] = join8(*(const v8h*)p, *(const v8h*)(p + 16));
        }
        // Branchless WGP-scope prefetch of next K tile; 32 lanes x 2 cover all 64 rows.
        __builtin_prefetch(A  + (size_t)(m0 + lane) * NN_ + k0 + 32, 0, 3);
        __builtin_prefetch(A  + (size_t)(m0 + 32 + lane) * NN_ + k0 + 32, 0, 3);
        __builtin_prefetch(Bt + (size_t)(c0 + lane) * NN_ + k0 + 32, 0, 3);
        __builtin_prefetch(Bt + (size_t)(c0 + 32 + lane) * NN_ + k0 + 32, 0, 3);
#pragma unroll
        for (int t = 0; t < 4; ++t)
#pragma unroll
            for (int u = 0; u < 4; ++u)
                acc[t][u] = __builtin_amdgcn_wmma_f32_16x16x32_f16(
                    false, af[t], false, bf[u], (short)0, acc[t][u], false, false);
    }
    // Transposed epilogue: per (t,u) tile each lane writes 8 consecutive n's of one c row.
#pragma unroll
    for (int t = 0; t < 4; ++t)
#pragma unroll
        for (int u = 0; u < 4; ++u) {
            int ccol = c0 + u * 16 + lh;
            int nrow = m0 + t * 16 + hp * 8;
            size_t off = (size_t)ccol * NN_ + nrow;
            v8f v = acc[t][u];
            if (beta != 0.0f) {
                v8f e = __builtin_convertvector(*(const v8h*)(Et + off), v8f);
#pragma unroll
                for (int i = 0; i < 8; ++i) v[i] = alpha * v[i] + beta * e[i];
            }
            *(v8h*)(Ct + off) = __builtin_convertvector(v, v8h);
        }
}

// ---------------- Feature GEMM with fused activation / GRU gating ----------------
// Af : (ROWS_ x KKP_) f16 row-major ;  Wt : (O x KKP_) f16 (transposed weights)
// mode 0: value[row*128+col] = sigmoid(acc + bias)          (O = 128, MW = 2)
// mode 1: out[row*64+col] = u*hx + (1-u)*tanh(acc + bias)   (O = 64,  MW = 4)
__global__ __launch_bounds__(256) void wmma_feat_gemm(const h_t* __restrict__ Af,
                                                      const h_t* __restrict__ Wt,
                                                      const float* __restrict__ bias,
                                                      float* __restrict__ value,
                                                      const float* __restrict__ hx,
                                                      float* __restrict__ out,
                                                      int MW, int mode) {
    const int lane = threadIdx.x & 31;
    const int lh   = lane & 15;
    const int hp   = lane >> 4;
    const int wid  = threadIdx.x >> 5;
    const int wm   = wid % MW;
    const int wo   = wid / MW;
    const int m0   = blockIdx.x * (MW * 64) + wm * 64;
    const int o0   = wo * 32;

    v8f acc[4][2] = {};
    for (int k0 = 0; k0 < KKP_; k0 += 32) {
        v16h af[4];
#pragma unroll
        for (int t = 0; t < 4; ++t) {
            const h_t* p = Af + (size_t)(m0 + t * 16 + lh) * KKP_ + k0 + hp * 8;
            af[t] = join8(*(const v8h*)p, *(const v8h*)(p + 16));
        }
        v16h bf[2];
#pragma unroll
        for (int u = 0; u < 2; ++u) {
            const h_t* p = Wt + (size_t)(o0 + u * 16 + lh) * KKP_ + k0 + hp * 8;
            bf[u] = join8(*(const v8h*)p, *(const v8h*)(p + 16));
        }
#pragma unroll
        for (int t = 0; t < 4; ++t)
#pragma unroll
            for (int u = 0; u < 2; ++u)
                acc[t][u] = __builtin_amdgcn_wmma_f32_16x16x32_f16(
                    false, af[t], false, bf[u], (short)0, acc[t][u], false, false);
    }
#pragma unroll
    for (int t = 0; t < 4; ++t)
#pragma unroll
        for (int u = 0; u < 2; ++u) {
            int col = o0 + u * 16 + lh;
            float bcol = bias[col];
#pragma unroll
            for (int v = 0; v < 8; ++v) {
                size_t row = (size_t)(m0 + t * 16 + hp * 8 + v);
                float x = acc[t][u][v] + bcol;
                if (mode == 0) {
                    value[row * 128 + col] = 1.0f / (1.0f + expf(-x));
                } else {
                    float c  = tanhf(x);
                    float ug = value[row * 128 + 64 + col];
                    float h  = hx[row * UU_ + col];
                    out[row * UU_ + col] = ug * h + (1.0f - ug) * c;
                }
            }
        }
}

// ---------------- Host launcher ----------------
extern "C" void kernel_launch(void* const* d_in, const int* in_sizes, int n_in,
                              void* d_out, int out_size, void* d_ws, size_t ws_size,
                              hipStream_t stream) {
    const float* inputs   = (const float*)d_in[0];   // (64, 8192)
    const float* hx       = (const float*)d_in[1];   // (64, 262144)
    const float* supports = (const float*)d_in[2];   // (2, 4096, 4096)
    const float* ru_w     = (const float*)d_in[3];   // (330, 128)
    const float* ru_b     = (const float*)d_in[4];   // (128,)
    const float* gc_w     = (const float*)d_in[5];   // (330, 64)
    const float* gc_b     = (const float*)d_in[6];   // (64,)
    float* outp           = (float*)d_out;           // (64, 262144)
    (void)in_sizes; (void)n_in; (void)out_size; (void)ws_size;

    const size_t NNsq = (size_t)NN_ * NN_;           // 16,777,216
    const size_t FBN  = (size_t)FBL_ * NN_;          // 17,301,504

    size_t off = 0;
    auto carve = [&](size_t bytes) {
        void* p = (char*)d_ws + off;
        off += (bytes + 255) & ~(size_t)255;
        return p;
    };
    h_t*   S_h   = (h_t*)  carve(2 * NNsq * sizeof(h_t));            //  67 MB
    h_t*   xsT   = (h_t*)  carve(5 * FBN * sizeof(h_t));             // 173 MB
    h_t*   Af    = (h_t*)  carve((size_t)ROWS_ * KKP_ * sizeof(h_t));// 184 MB
    h_t*   Wt1   = (h_t*)  carve(128 * KKP_ * sizeof(h_t));
    h_t*   Wt2   = (h_t*)  carve(64  * KKP_ * sizeof(h_t));
    float* value = (float*)carve((size_t)ROWS_ * 128 * sizeof(float)); // 134 MB

    // One-time f16 staging
    pack_supports<<<32768, 256, 0, stream>>>(supports, S_h, (int)(2 * NNsq / 4));
    pack_wt<<<176, 256, 0, stream>>>(ru_w, Wt1, 128);
    pack_wt<<<88,  256, 0, stream>>>(gc_w, Wt2, 64);

    const dim3 gBig(NN_ / 256, FBL_ / 128);   // 16 x 33
    for (int stage = 0; stage < 2; ++stage) {
        pack_x0<<<67584, 256, 0, stream>>>(inputs, hx, value, stage, xsT);
        for (int s = 0; s < 2; ++s) {
            const h_t* S  = S_h + (size_t)s * NNsq;
            h_t* x1 = xsT + (size_t)(1 + 2 * s) * FBN;
            h_t* x2 = xsT + (size_t)(2 + 2 * s) * FBN;
            // x1 = S @ x0
            wmma_sx_gemm<<<gBig, 256, 0, stream>>>(S, xsT, nullptr, x1, 1.0f, 0.0f);
            // x2 = 2 * (S @ x1) - x0
            wmma_sx_gemm<<<gBig, 256, 0, stream>>>(S, x1, xsT, x2, 2.0f, -1.0f);
        }
        pack_afeat<<<360448, 256, 0, stream>>>(xsT, Af);
        if (stage == 0) {
            // value = sigmoid(Afeat @ ru_w + ru_b)   -> r,u gates
            wmma_feat_gemm<<<2048, 256, 0, stream>>>(Af, Wt1, ru_b, value,
                                                     nullptr, nullptr, 2, 0);
        } else {
            // out = u*hx + (1-u)*tanh(Afeat @ gc_w + gc_b)
            wmma_feat_gemm<<<1024, 256, 0, stream>>>(Af, Wt2, gc_b, value,
                                                     hx, outp, 4, 1);
        }
    }
}
